// TopKGate_9964324127039
// MI455X (gfx1250) — matrix operand order repaired and checked by reading, not compiled, not run on hardware
//
#include <hip/hip_runtime.h>
#include <hip/hip_bf16.h>

// ---------------------------------------------------------------------------
// TopK (K=2) MoE gate for MI455X / gfx1250.
//   logits = h[16384,2048] @ W[64,2048]^T  via v_wmma_f32_16x16x32_bf16
//   top-2 + softmax over the two winners.
// One wave = 16 tokens x 64 experts. 8 waves / block, 128 blocks.
// W is converted f32->bf16 once per block into 257KB of LDS (CDNA5 320KB WGP LDS).
// h stream is register double-buffered across k-steps for memory-level
// parallelism, with clamped (always-valid) global_prefetch ahead of the stream.
// ---------------------------------------------------------------------------

typedef __attribute__((ext_vector_type(16))) __bf16 v16bf;
typedef __attribute__((ext_vector_type(8)))  __bf16 v8bf;
typedef __attribute__((ext_vector_type(4)))  __bf16 v4bf;
typedef __attribute__((ext_vector_type(8)))  float  v8f;

#define DMODEL     2048
#define NEXPERTS   64
#define TOKENS     16384
#define LDS_STRIDE 2056                      // 2048 + 8 halves (16B pad): bank-conflict-free b128
#define LDS_BYTES  (NEXPERTS * LDS_STRIDE * 2)

__device__ __forceinline__ bool rank_before(float va, int ia, float vb, int ib) {
    // "a ranks before b": larger value wins, ties broken by lower index (lax.top_k)
    return (va > vb) || (va == vb && ia < ib);
}

__device__ __forceinline__ void merge_top2(float& v1, int& i1, float& v2, int& i2,
                                           float w1, int j1, float w2, int j2) {
    // merge two descending pairs into the overall top-2
    if (rank_before(w1, j1, v1, i1)) {
        if (rank_before(v1, i1, w2, j2)) { v2 = v1; i2 = i1; }
        else                             { v2 = w2; i2 = j2; }
        v1 = w1; i1 = j1;
    } else if (rank_before(w1, j1, v2, i2)) {
        v2 = w1; i2 = j1;
    }
}

// One K=32 step: pack the A fragment from 16 f32 values, load 4 B fragments
// from LDS, issue 4 WMMAs (64 experts).
__device__ __forceinline__ void wmma_step(float4 a0, float4 a1, float4 a2, float4 a3,
                                          const __bf16* __restrict__ pB,
                                          v8f acc[4]) {
    v16bf A;
    A[0]  = (__bf16)a0.x; A[1]  = (__bf16)a0.y; A[2]  = (__bf16)a0.z; A[3]  = (__bf16)a0.w;
    A[4]  = (__bf16)a1.x; A[5]  = (__bf16)a1.y; A[6]  = (__bf16)a1.z; A[7]  = (__bf16)a1.w;
    A[8]  = (__bf16)a2.x; A[9]  = (__bf16)a2.y; A[10] = (__bf16)a2.z; A[11] = (__bf16)a2.w;
    A[12] = (__bf16)a3.x; A[13] = (__bf16)a3.y; A[14] = (__bf16)a3.z; A[15] = (__bf16)a3.w;

    #pragma unroll
    for (int et = 0; et < 4; ++et) {
        const __bf16* pBe = pB + et * 16 * LDS_STRIDE;         // experts et*16 + n
        const v8bf b0 = *reinterpret_cast<const v8bf*>(pBe);
        const v8bf b1 = *reinterpret_cast<const v8bf*>(pBe + 16);
        const v16bf B = __builtin_shufflevector(b0, b1,
            0, 1, 2, 3, 4, 5, 6, 7, 8, 9, 10, 11, 12, 13, 14, 15);
        acc[et] = __builtin_amdgcn_wmma_f32_16x16x32_bf16(
            false, A, false, B, (short)0, acc[et], false, false);
    }
}

__global__ __launch_bounds__(256)
void topk_gate_kernel(const float* __restrict__ h,
                      const float* __restrict__ W,
                      float* __restrict__ gates,
                      int* __restrict__ idxOut) {
    extern __shared__ __align__(16) __bf16 Wlds[];   // [NEXPERTS][LDS_STRIDE]

    // ---------------- Phase 1: stage W as bf16 into LDS (cooperative) -------
    {
        const int total = NEXPERTS * DMODEL;           // 131072 floats
        for (int i = threadIdx.x * 4; i < total; i += blockDim.x * 4) {
            const int e = i >> 11;                     // /2048
            const int k = i & (DMODEL - 1);
            const float4 w4 = *reinterpret_cast<const float4*>(W + i);
            v4bf p;
            p[0] = (__bf16)w4.x; p[1] = (__bf16)w4.y;
            p[2] = (__bf16)w4.z; p[3] = (__bf16)w4.w;
            *reinterpret_cast<v4bf*>(&Wlds[e * LDS_STRIDE + k]) = p;
        }
    }
    __syncthreads();

    // ---------------- Phase 2: 16x64 logits tile per wave via WMMA ----------
    const int lane = threadIdx.x & 31;
    const int wave = threadIdx.x >> 5;
    const int n    = lane & 15;          // A row (token-in-tile) == B column (expert-in-tile)
    const int half = lane >> 4;          // which K-half this lane supplies

    const long t0   = (long)blockIdx.x * 128 + (long)wave * 16;   // first token of tile
    const float* hRow = h + (t0 + n) * DMODEL;

    v8f acc[4] = {};                     // 4 expert tiles of 16 -> 64 experts

    // A fragment layout (16-bit A 16x32): lane supplies
    //   K = kb + half*8 + 0..7   and   K = kb + 16 + half*8 + 0..7
    const float* pA0 = hRow + half * 8;
    float4 a0 = *reinterpret_cast<const float4*>(pA0);
    float4 a1 = *reinterpret_cast<const float4*>(pA0 + 4);
    float4 a2 = *reinterpret_cast<const float4*>(pA0 + 16);
    float4 a3 = *reinterpret_cast<const float4*>(pA0 + 20);

    #pragma unroll 2
    for (int kb = 0; kb < DMODEL - 32; kb += 32) {
        // ---- software pipeline: fetch NEXT A chunk while computing current
        const float* pN = hRow + (kb + 32) + half * 8;
        const float4 n0 = *reinterpret_cast<const float4*>(pN);
        const float4 n1 = *reinterpret_cast<const float4*>(pN + 4);
        const float4 n2 = *reinterpret_cast<const float4*>(pN + 16);
        const float4 n3 = *reinterpret_cast<const float4*>(pN + 20);

        // branch-free prefetch ~9 steps ahead; address clamped into the row
        const int kpf = (kb + 320 < DMODEL) ? (kb + 320 + half * 8) : (DMODEL - 1);
        __builtin_prefetch(hRow + kpf, 0, 3);

        wmma_step(a0, a1, a2, a3, &Wlds[n * LDS_STRIDE + kb + half * 8], acc);

        a0 = n0; a1 = n1; a2 = n2; a3 = n3;
    }
    // peeled final k-step
    wmma_step(a0, a1, a2, a3, &Wlds[n * LDS_STRIDE + (DMODEL - 32) + half * 8], acc);

    // ---------------- Phase 3: per-token top-2 + softmax --------------------
    // C layout: VGPR r, lane -> token M = r + 8*half, expert = et*16 + n.
    #pragma unroll
    for (int r = 0; r < 8; ++r) {
        // local top-2 across the 4 expert tiles held by this lane
        float v1, v2; int i1, i2;
        {
            const float a = acc[0][r], b = acc[1][r];
            if (rank_before(b, n + 16, a, n)) { v1 = b; i1 = n + 16; v2 = a; i2 = n; }
            else                              { v1 = a; i1 = n;      v2 = b; i2 = n + 16; }
            #pragma unroll
            for (int j = 2; j < 4; ++j) {
                const float c = acc[j][r]; const int ic = n + 16 * j;
                if (rank_before(c, ic, v1, i1)) { v2 = v1; i2 = i1; v1 = c; i1 = ic; }
                else if (rank_before(c, ic, v2, i2)) { v2 = c; i2 = ic; }
            }
        }
        // butterfly all-reduce across the 16 lanes of this half (wave32 shuffles)
        #pragma unroll
        for (int mask = 1; mask < 16; mask <<= 1) {
            const float w1 = __shfl_xor(v1, mask, 32);
            const int   j1 = __shfl_xor(i1, mask, 32);
            const float w2 = __shfl_xor(v2, mask, 32);
            const int   j2 = __shfl_xor(i2, mask, 32);
            merge_top2(v1, i1, v2, i2, w1, j1, w2, j2);
        }
        // one writer per token: lane with n == r (both halves write distinct tokens)
        if (n == r) {
            const long t = t0 + r + 8 * half;
            const float g1 = 1.0f / (1.0f + __expf(v2 - v1));   // softmax of [v1 >= v2]
            const float g2 = 1.0f - g1;
            *reinterpret_cast<float2*>(gates + t * 2) = make_float2(g1, g2);
            *reinterpret_cast<int2*>(idxOut + t * 2)  = make_int2(i1, i2);
        }
    }
}

extern "C" void kernel_launch(void* const* d_in, const int* in_sizes, int n_in,
                              void* d_out, int out_size, void* d_ws, size_t ws_size,
                              hipStream_t stream) {
    (void)in_sizes; (void)n_in; (void)out_size; (void)d_ws; (void)ws_size;
    const float* h = (const float*)d_in[0];       // [4,4096,2048] f32
    const float* W = (const float*)d_in[1];       // [64,2048]     f32
    float* gates = (float*)d_out;                 // [4,4096,2]    f32
    int*   idx   = (int*)d_out + TOKENS * 2;      // [4,4096,2]    i32 (concatenated)

    const dim3 grid(TOKENS / 128);                // 8 waves * 16 tokens per block
    const dim3 block(256);
    topk_gate_kernel<<<grid, block, LDS_BYTES, stream>>>(h, W, gates, idx);
}
